// HSTULayer_67680094650963
// MI455X (gfx1250) — matrix operand order repaired and checked by reading, not compiled, not run on hardware
//
#include <hip/hip_runtime.h>

// ---------------------------------------------------------------------------
// HSTU layer for MI455X (gfx1250), bf16 WMMA pipeline.
// B=4, N=2048, D=512, H=8, DH=64, T=8192
// Round 4: force wave-uniform tile indices into SGPRs via readfirstlane so
// the epilogue section-routing compiles to scalar branches (s_cbranch) instead
// of saveexec/v_cmpx divergence, and tile bases live in SGPRs.
// ---------------------------------------------------------------------------

#define TB 8192      // tokens
#define DD 512       // hidden
#define NN 2048      // seqlen
#define HH 8         // heads
#define DHH 64       // head dim

typedef __attribute__((ext_vector_type(16))) __bf16 v16bf;
typedef __attribute__((ext_vector_type(8)))  float  v8f;
typedef __attribute__((ext_vector_type(8)))  unsigned int v8u;

__device__ __forceinline__ v8f zero8() {
    v8f z;
#pragma unroll
    for (int i = 0; i < 8; ++i) z[i] = 0.f;
    return z;
}

__device__ __forceinline__ unsigned short f2bf(float f) {
    unsigned int u = __builtin_bit_cast(unsigned int, f);
    unsigned int r = u + 0x7fffu + ((u >> 16) & 1u);   // round-to-nearest-even
    return (unsigned short)(r >> 16);
}

__device__ __forceinline__ unsigned int pack2bf(float lo, float hi) {
    return (unsigned int)f2bf(lo) | ((unsigned int)f2bf(hi) << 16);
}

// silu(x) = x * sigmoid(x); v_exp_f32 + v_rcp_f32 fast path (bf16 precision).
__device__ __forceinline__ float siluf(float x) {
    return x * __builtin_amdgcn_rcpf(1.f + __expf(-x));
}

// Wave-uniform scalarization helper.
__device__ __forceinline__ int uni(int v) {
    return __builtin_amdgcn_readfirstlane(v);
}

// Load 2x16-byte chunks (8 bf16 each) into one 16-element bf16 fragment.
__device__ __forceinline__ v8u ld_frag(const unsigned short* p0, const unsigned short* p1) {
    uint4 a = *(const uint4*)p0;
    uint4 b = *(const uint4*)p1;
    v8u r;
    r[0]=a.x; r[1]=a.y; r[2]=a.z; r[3]=a.w;
    r[4]=b.x; r[5]=b.y; r[6]=b.z; r[7]=b.w;
    return r;
}

__device__ __forceinline__ v8f wmma_bf16(v8u a, v8u b, v8f c) {
    return __builtin_amdgcn_wmma_f32_16x16x32_bf16(
        false, __builtin_bit_cast(v16bf, a),
        false, __builtin_bit_cast(v16bf, b),
        (short)0, c, false, false);
}

__device__ __forceinline__ float wred(float v) {
#pragma unroll
    for (int m = 16; m > 0; m >>= 1) v += __shfl_xor(v, m, 32);
    return v;
}

// ---------------------------------------------------------------------------
// Kernel 1: LayerNorm1 -> bf16 row-major [T][512]. One wave per row.
// ---------------------------------------------------------------------------
__global__ void ln1_kernel(const float* __restrict__ x, const float* __restrict__ g,
                           const float* __restrict__ b, unsigned short* __restrict__ out) {
    const int tid  = blockIdx.x * blockDim.x + threadIdx.x;
    const int row  = uni(tid >> 5);
    const int lane = threadIdx.x & 31;

    const float* xr = x + (size_t)row * DD + lane * 16;
    float v[16];
#pragma unroll
    for (int i = 0; i < 4; ++i) {
        float4 t = ((const float4*)xr)[i];
        v[i*4+0]=t.x; v[i*4+1]=t.y; v[i*4+2]=t.z; v[i*4+3]=t.w;
    }
    float s = 0.f, sq = 0.f;
#pragma unroll
    for (int i = 0; i < 16; ++i) { s += v[i]; sq += v[i]*v[i]; }
    s = wred(s); sq = wred(sq);
    const float mean = s * (1.f/DD);
    const float var  = sq * (1.f/DD) - mean*mean;
    const float rstd = rsqrtf(var + 1e-5f);

    unsigned int d[8];
#pragma unroll
    for (int j = 0; j < 8; ++j) {
        int c0 = lane*16 + 2*j;
        float a0 = (v[2*j]   - mean) * rstd * g[c0]   + b[c0];
        float a1 = (v[2*j+1] - mean) * rstd * g[c0+1] + b[c0+1];
        d[j] = pack2bf(a0, a1);
    }
    uint4* dst = (uint4*)(out + (size_t)row * DD + lane * 16);
    dst[0] = make_uint4(d[0],d[1],d[2],d[3]);
    dst[1] = make_uint4(d[4],d[5],d[6],d[7]);
}

// ---------------------------------------------------------------------------
// Kernel 2: pack weight f32 [K][C] -> bf16 B-fragment tiles.
// Tile (nt16, kt32): lane l holds col = nt16*16 + l%16, k = kt32*32 + (l/16)*16 + j.
// One wave per tile; tile index == wave index.
// ---------------------------------------------------------------------------
__global__ void pack_w(const float* __restrict__ W, unsigned short* __restrict__ out,
                       int K, int C) {
    const int tid  = blockIdx.x * blockDim.x + threadIdx.x;
    const int wave = uni(tid >> 5);
    const int lane = threadIdx.x & 31;
    const int l16  = lane & 15;
    const int half = lane >> 4;
    const int tiles_k = K >> 5;
    const int nt16 = wave / tiles_k;
    const int kt   = wave % tiles_k;
    if (nt16 >= (C >> 4)) return;

    const int col = nt16*16 + l16;
    const int kb  = kt*32 + half*16;
    unsigned int d[8];
#pragma unroll
    for (int i = 0; i < 8; ++i) {
        float lo = W[(size_t)(kb + 2*i  ) * C + col];
        float hi = W[(size_t)(kb + 2*i+1) * C + col];
        d[i] = pack2bf(lo, hi);
    }
    uint4* dst = (uint4*)(out + (size_t)wave * 512 + lane * 16);
    dst[0] = make_uint4(d[0],d[1],d[2],d[3]);
    dst[1] = make_uint4(d[4],d[5],d[6],d[7]);
}

// ---------------------------------------------------------------------------
// Kernel 3: GEMM1 [T,512]x[512,2048] bf16 WMMA + SiLU + split to u/q/k/vT.
// One wave computes a 32x64 tile (8 accum frags -> no spills).
// ---------------------------------------------------------------------------
__global__ void __launch_bounds__(256, 1)
gemm1_silu_split(const unsigned short* __restrict__ xn,
                 const unsigned short* __restrict__ wpk,
                 float* __restrict__ u,
                 unsigned short* __restrict__ q,
                 unsigned short* __restrict__ k,
                 unsigned short* __restrict__ vT) {
    const int tid  = blockIdx.x * blockDim.x + threadIdx.x;
    const int wave = uni(tid >> 5);
    const int lane = threadIdx.x & 31;
    const int l16  = lane & 15;
    const int half = lane >> 4;
    const int mt = wave / 32;   // 0..255 : 32-row tile
    const int nt = wave % 32;   // 0..31  : 64-col tile

    v8f acc[8];
#pragma unroll
    for (int i = 0; i < 8; ++i) acc[i] = zero8();

    const unsigned short* arow[2];
#pragma unroll
    for (int sm = 0; sm < 2; ++sm)
        arow[sm] = xn + (size_t)(mt*32 + sm*16 + l16) * DD;

    for (int kt = 0; kt < 16; ++kt) {
        v8u A[2], Bf[4];
#pragma unroll
        for (int sm = 0; sm < 2; ++sm) {
            const unsigned short* p = arow[sm] + kt*32 + half*16;
            A[sm] = ld_frag(p, p + 8);
        }
#pragma unroll
        for (int sn = 0; sn < 4; ++sn) {
            const unsigned short* p = wpk + ((size_t)((nt*4 + sn)*16 + kt)) * 512 + lane*16;
            Bf[sn] = ld_frag(p, p + 8);
        }
#pragma unroll
        for (int sm = 0; sm < 2; ++sm)
#pragma unroll
            for (int sn = 0; sn < 4; ++sn)
                acc[sm*4+sn] = wmma_bf16(A[sm], Bf[sn], acc[sm*4+sn]);
    }

    const int h   = nt >> 2;   // head  (wave-uniform, scalar)
    const int sec = nt & 3;    // 0=user 1=value 2=query 3=key (scalar)
#pragma unroll
    for (int sm = 0; sm < 2; ++sm) {
#pragma unroll
        for (int sn = 0; sn < 4; ++sn) {
            v8f a = acc[sm*4+sn];
#pragma unroll
            for (int r = 0; r < 8; ++r) {
                float val = siluf(a[r]);
                int t = mt*32 + sm*16 + r + half*8;  // global token row
                int n = sn*16 + l16;                 // dh within section (0..63)
                if (sec == 0) {
                    u[(size_t)t * DD + h*DHH + n] = val;
                } else {
                    int bb = t >> 11, s = t & (NN - 1);
                    unsigned short bv = f2bf(val);
                    if (sec == 1)
                        vT[(((size_t)(bb*HH + h) * DHH + n) << 11) + s] = bv;
                    else if (sec == 2)
                        q[(((size_t)(bb*HH + h) << 11) + s) * DHH + n] = bv;
                    else
                        k[(((size_t)(bb*HH + h) << 11) + s) * DHH + n] = bv;
                }
            }
        }
    }
}

// ---------------------------------------------------------------------------
// Kernel 4: HSTU attention. One wave per (b, h, 16-query tile).
// Scores via WMMA (K=DH=64 -> 2 steps), silu/scale/causal in regs,
// p re-layout via LDS bounce, then p@v WMMA accumulating 16x64 output.
// ---------------------------------------------------------------------------
__global__ void __launch_bounds__(256, 1)
hstu_attn(const unsigned short* __restrict__ q,
          const unsigned short* __restrict__ k,
          const unsigned short* __restrict__ vT,
          float* __restrict__ attn,
          const int* __restrict__ scale) {
    __shared__ __align__(16) unsigned short ps[8][16*32];  // per-wave 16x32 bf16 p tile

    const int tid  = blockIdx.x * blockDim.x + threadIdx.x;
    const int wave = uni(tid >> 5);
    const int wv   = uni(threadIdx.x >> 5);
    const int lane = threadIdx.x & 31;
    const int l16  = lane & 15;
    const int half = lane >> 4;

    const int mt = wave & 127;         // q tile (16 rows)  (scalar)
    const int h  = (wave >> 7) & 7;    // (scalar)
    const int b  = wave >> 10;         // (scalar)
    const float inv = 1.f / (float)(*scale);
    const size_t bh = (size_t)(b*HH + h);

    // q A-fragments for both K-steps (dh 0..31, 32..63)
    const unsigned short* qp = q + (bh*NN + mt*16 + l16) * DHH;
    v8u qA[2];
#pragma unroll
    for (int s = 0; s < 2; ++s) {
        const unsigned short* p = qp + s*32 + half*16;
        qA[s] = ld_frag(p, p + 8);
    }

    v8f o[4];
#pragma unroll
    for (int i = 0; i < 4; ++i) o[i] = zero8();

    const int nkc = (mt*16 + 15) / 32 + 1;   // causal 32-key chunks (scalar)
    for (int kc = 0; kc < nkc; ++kc) {
        // --- scores: two 16x16 tiles (keys kc*32+x*16 ..)
        v8f S[2];
#pragma unroll
        for (int x = 0; x < 2; ++x) {
            S[x] = zero8();
            const int key = kc*32 + x*16 + l16;
            const unsigned short* kp = k + (bh*NN + key) * DHH;
#pragma unroll
            for (int s = 0; s < 2; ++s) {
                const unsigned short* p = kp + s*32 + half*16;
                v8u Bf = ld_frag(p, p + 8);
                S[x] = wmma_bf16(qA[s], Bf, S[x]);
            }
        }
        // --- silu / scale / causal mask -> LDS bf16 p tile [16][32]
#pragma unroll
        for (int x = 0; x < 2; ++x) {
#pragma unroll
            for (int r = 0; r < 8; ++r) {
                int m  = r + half*8;
                int qi = mt*16 + m;
                int kj = kc*32 + x*16 + l16;
                float pv = (kj <= qi) ? siluf(S[x][r]) * inv : 0.f;
                ps[wv][m*32 + x*16 + l16] = f2bf(pv);
            }
        }
        // --- p A-fragment from LDS (two ds_load_b128 per lane)
        const unsigned short* pr = &ps[wv][l16*32 + half*16];
        v8u pA = ld_frag(pr, pr + 8);
        // --- p @ v : 4 dh sub-tiles
#pragma unroll
        for (int sn = 0; sn < 4; ++sn) {
            const unsigned short* vp = vT + (bh*DHH + sn*16 + l16) * NN + kc*32 + half*16;
            v8u Bf = ld_frag(vp, vp + 8);
            o[sn] = wmma_bf16(pA, Bf, o[sn]);
        }
    }

    // store attn f32 [T][512]
    const int t0 = b*NN + mt*16;
#pragma unroll
    for (int sn = 0; sn < 4; ++sn)
#pragma unroll
        for (int r = 0; r < 8; ++r) {
            int m = r + half*8;
            attn[(size_t)(t0 + m) * DD + h*DHH + sn*16 + l16] = o[sn][r];
        }
}

// ---------------------------------------------------------------------------
// Kernel 5: LayerNorm2 + user gate -> bf16. One wave per row.
// ---------------------------------------------------------------------------
__global__ void ln2_gate(const float* __restrict__ a, const float* __restrict__ g,
                         const float* __restrict__ b, const float* __restrict__ u,
                         unsigned short* __restrict__ out) {
    const int tid  = blockIdx.x * blockDim.x + threadIdx.x;
    const int row  = uni(tid >> 5);
    const int lane = threadIdx.x & 31;

    const float* ar = a + (size_t)row * DD + lane * 16;
    float v[16];
#pragma unroll
    for (int i = 0; i < 4; ++i) {
        float4 t = ((const float4*)ar)[i];
        v[i*4+0]=t.x; v[i*4+1]=t.y; v[i*4+2]=t.z; v[i*4+3]=t.w;
    }
    float s = 0.f, sq = 0.f;
#pragma unroll
    for (int i = 0; i < 16; ++i) { s += v[i]; sq += v[i]*v[i]; }
    s = wred(s); sq = wred(sq);
    const float mean = s * (1.f/DD);
    const float var  = sq * (1.f/DD) - mean*mean;
    const float rstd = rsqrtf(var + 1e-5f);

    const float* ur = u + (size_t)row * DD + lane * 16;
    unsigned int d[8];
#pragma unroll
    for (int j = 0; j < 8; ++j) {
        int c0 = lane*16 + 2*j;
        float a0 = ((v[2*j]   - mean) * rstd * g[c0]   + b[c0])   * ur[2*j];
        float a1 = ((v[2*j+1] - mean) * rstd * g[c0+1] + b[c0+1]) * ur[2*j+1];
        d[j] = pack2bf(a0, a1);
    }
    uint4* dst = (uint4*)(out + (size_t)row * DD + lane * 16);
    dst[0] = make_uint4(d[0],d[1],d[2],d[3]);
    dst[1] = make_uint4(d[4],d[5],d[6],d[7]);
}

// ---------------------------------------------------------------------------
// Kernel 6: GEMM2 [T,512]x[512,512] bf16 WMMA + residual -> f32 out.
// 32x64 tile per wave.
// ---------------------------------------------------------------------------
__global__ void __launch_bounds__(256, 1)
gemm2_residual(const unsigned short* __restrict__ gbf,
               const unsigned short* __restrict__ wpk,
               const float* __restrict__ x,
               float* __restrict__ out) {
    const int tid  = blockIdx.x * blockDim.x + threadIdx.x;
    const int wave = uni(tid >> 5);
    const int lane = threadIdx.x & 31;
    const int l16  = lane & 15;
    const int half = lane >> 4;
    const int mt = wave / 8;   // 0..255 (scalar)
    const int nt = wave % 8;   // 0..7   (scalar)

    v8f acc[8];
#pragma unroll
    for (int i = 0; i < 8; ++i) acc[i] = zero8();

    const unsigned short* arow[2];
#pragma unroll
    for (int sm = 0; sm < 2; ++sm)
        arow[sm] = gbf + (size_t)(mt*32 + sm*16 + l16) * DD;

    for (int kt = 0; kt < 16; ++kt) {
        v8u A[2], Bf[4];
#pragma unroll
        for (int sm = 0; sm < 2; ++sm) {
            const unsigned short* p = arow[sm] + kt*32 + half*16;
            A[sm] = ld_frag(p, p + 8);
        }
#pragma unroll
        for (int sn = 0; sn < 4; ++sn) {
            const unsigned short* p = wpk + ((size_t)((nt*4 + sn)*16 + kt)) * 512 + lane*16;
            Bf[sn] = ld_frag(p, p + 8);
        }
#pragma unroll
        for (int sm = 0; sm < 2; ++sm)
#pragma unroll
            for (int sn = 0; sn < 4; ++sn)
                acc[sm*4+sn] = wmma_bf16(A[sm], Bf[sn], acc[sm*4+sn]);
    }

#pragma unroll
    for (int sm = 0; sm < 2; ++sm)
#pragma unroll
        for (int sn = 0; sn < 4; ++sn) {
            v8f a = acc[sm*4+sn];
#pragma unroll
            for (int r = 0; r < 8; ++r) {
                int t = mt*32 + sm*16 + r + half*8;
                int c = nt*64 + sn*16 + l16;
                out[(size_t)t * DD + c] = a[r] + x[(size_t)t * DD + c];
            }
        }
}

// ---------------------------------------------------------------------------
// Host launcher
// ---------------------------------------------------------------------------
extern "C" void kernel_launch(void* const* d_in, const int* in_sizes, int n_in,
                              void* d_out, int out_size, void* d_ws, size_t ws_size,
                              hipStream_t stream) {
    (void)in_sizes; (void)n_in; (void)out_size; (void)ws_size;

    const float* x      = (const float*)d_in[0];
    const float* ln1g   = (const float*)d_in[1];
    const float* ln1b   = (const float*)d_in[2];
    const float* w_uvqk = (const float*)d_in[3];
    const float* ln2g   = (const float*)d_in[4];
    const float* ln2b   = (const float*)d_in[5];
    const float* w_proj = (const float*)d_in[6];
    const int*   scal   = (const int*)d_in[7];
    float* out = (float*)d_out;

    char* ws = (char*)d_ws;
    size_t off = 0;
    auto carve = [&](size_t bytes) -> void* {
        void* p = ws + off;
        off += (bytes + 255) & ~(size_t)255;
        return p;
    };
    unsigned short* xn   = (unsigned short*)carve((size_t)TB * DD * 2);          // LN1(x) bf16
    unsigned short* wuv  = (unsigned short*)carve((size_t)DD * 2048 * 2);        // packed w_uvqk
    unsigned short* wpj  = (unsigned short*)carve((size_t)DD * DD * 2);          // packed w_proj
    float*          u    = (float*)carve((size_t)TB * DD * 4);                   // user gate f32
    unsigned short* qb   = (unsigned short*)carve((size_t)TB * DD * 2);          // q bf16 [b,h,n,dh]
    unsigned short* kb   = (unsigned short*)carve((size_t)TB * DD * 2);          // k bf16 [b,h,n,dh]
    unsigned short* vT   = (unsigned short*)carve((size_t)TB * DD * 2);          // v bf16 [b,h,dh,n]
    float*          attn = (float*)carve((size_t)TB * DD * 4);                   // attn f32
    unsigned short* gb   = (unsigned short*)carve((size_t)TB * DD * 2);          // gated bf16

    // 1. LN1  (8192 waves)
    ln1_kernel<<<TB/8, 256, 0, stream>>>(x, ln1g, ln1b, xn);
    // 2. weight packs: w_uvqk 2048 tiles, w_proj 512 tiles (8 waves / block)
    pack_w<<<2048/8, 256, 0, stream>>>(w_uvqk, wuv, DD, 2048);
    pack_w<<<512/8,  256, 0, stream>>>(w_proj, wpj, DD, DD);
    // 3. GEMM1 + SiLU + split  (256*32 = 8192 waves)
    gemm1_silu_split<<<8192/8, 256, 0, stream>>>(xn, wuv, u, qb, kb, vT);
    // 4. attention (4*8*128 = 4096 waves)
    hstu_attn<<<4096/8, 256, 0, stream>>>(qb, kb, vT, attn, scal);
    // 5. LN2 + gate (8192 waves)
    ln2_gate<<<TB/8, 256, 0, stream>>>(attn, ln2g, ln2b, u, gb);
    // 6. GEMM2 + residual (256*8 = 2048 waves)
    gemm2_residual<<<2048/8, 256, 0, stream>>>(gb, wpj, x, out);
}
